// ChamberOfSemanticResonance_33870112096743
// MI455X (gfx1250) — compile-verified, matrix operand order checked by reading, not run on hardware
//
#include <hip/hip_runtime.h>
#include <math.h>

// ---- CDNA5 WMMA types ----
typedef __attribute__((ext_vector_type(16))) __bf16 v16bf;
typedef __attribute__((ext_vector_type(8)))  float  v8f;

constexpr int T = 4096;   // B*S tokens
constexpr int D = 512;
constexpr int F = 2048;
constexpr int E = 8;
constexpr int TILE_M = 32;

union Frag {
  unsigned int u[8];
  uint4        q[2];
  v16bf        v;
};

// First K index held by VGPR v for the 16-bit A/B fragment layout (16x32 / 32x16).
// lanes 0-15 (half=0): V0:K0,1 V1:K2,3 V2:K4,5 V3:K6,7 V4:K16,17 ... V7:K22,23
// lanes 16-31 (half=1): +8
__device__ __forceinline__ int KB(int v, int half) {
  return (v < 4 ? 2 * v : 16 + 2 * (v - 4)) + 8 * half;
}

__device__ __forceinline__ unsigned int pack_bf16x2(float a, float b) {
  unsigned int ua = __float_as_uint(a), ub = __float_as_uint(b);
  ua = (ua + 0x7FFFu + ((ua >> 16) & 1u)) >> 16;   // RNE
  ub = (ub + 0x7FFFu + ((ub >> 16) & 1u)) >> 16;
  return (ub << 16) | (ua & 0xFFFFu);
}
__device__ __forceinline__ unsigned short bf16_of(float a) {
  unsigned int ua = __float_as_uint(a);
  return (unsigned short)((ua + 0x7FFFu + ((ua >> 16) & 1u)) >> 16);
}

// ---------------- kernel 1: normalized anchors (output a_n) ----------------
__global__ void anchors_norm_kernel(const float* __restrict__ A, float* __restrict__ an) {
  const int wv = threadIdx.x >> 5, lane = threadIdx.x & 31;   // wave = expert row
  const float* a = A + wv * D;
  float ss = 0.f;
  for (int k = lane; k < D; k += 32) { float v = a[k]; ss += v * v; }
#pragma unroll
  for (int m = 16; m; m >>= 1) ss += __shfl_xor(ss, m, 32);
  const float inv = 1.f / fmaxf(sqrtf(ss), 1e-8f);
  for (int k = lane; k < D; k += 32) an[wv * D + k] = a[k] * inv;
}

// ---------------- kernel 2: cosine scores + top-2 + softmax gates ----------------
__global__ void route_kernel(const float* __restrict__ x, const float* __restrict__ an,
                             float* __restrict__ scores, float* __restrict__ topkf,
                             int* __restrict__ eid, float* __restrict__ gval) {
  const int wv = threadIdx.x >> 5, lane = threadIdx.x & 31;
  const int t = blockIdx.x * 8 + wv;
  const float* xr = x + (size_t)t * D;
  float ss = 0.f, se[E] = {};
  for (int k = lane; k < D; k += 32) {
    const float xv = xr[k];
    ss += xv * xv;
#pragma unroll
    for (int e = 0; e < E; e++) se[e] += xv * an[e * D + k];
  }
#pragma unroll
  for (int m = 16; m; m >>= 1) {
    ss += __shfl_xor(ss, m, 32);
#pragma unroll
    for (int e = 0; e < E; e++) se[e] += __shfl_xor(se[e], m, 32);
  }
  if (lane == 0) {
    const float inv = 1.f / fmaxf(sqrtf(ss), 1e-8f);
    float sc[E];
#pragma unroll
    for (int e = 0; e < E; e++) { sc[e] = se[e] * inv; scores[t * E + e] = sc[e]; }
    int i1 = 0; float s1 = sc[0];
#pragma unroll
    for (int e = 1; e < E; e++) if (sc[e] > s1) { s1 = sc[e]; i1 = e; }
    int i2 = (i1 == 0) ? 1 : 0; float s2 = sc[i2];
#pragma unroll
    for (int e = 0; e < E; e++) if (e != i1 && sc[e] > s2) { s2 = sc[e]; i2 = e; }
    const float ex = expf(s2 - s1);            // s1 >= s2, stable softmax
    const float g1 = 1.f / (1.f + ex);
    const float g2 = ex / (1.f + ex);
    topkf[t * 2 + 0] = (float)i1;
    topkf[t * 2 + 1] = (float)i2;
    eid[t * 2 + 0] = i1;  gval[t * 2 + 0] = g1;
    eid[t * 2 + 1] = i2;  gval[t * 2 + 1] = g2;
  }
}

// -------- kernel 3: deterministic per-expert compaction (ballot prefix scan) --------
__global__ void bucket_kernel(const int* __restrict__ eid, int* __restrict__ list,
                              int* __restrict__ counts) {
  const int e = blockIdx.x, lane = threadIdx.x;
  int off = 0;
  for (int base = 0; base < 2 * T; base += 32) {
    const int i = base + lane;
    const bool m = (eid[i] == e);
    const unsigned long long bal = __ballot(m);
    const int pre = __popcll(bal & ((1ull << lane) - 1ull));
    if (m) list[e * T + off + pre] = i;       // slot id, in token order -> deterministic
    off += __popcll(bal);
  }
  if (lane == 0) counts[e] = off;
}

// -------- weight pre-swizzle: f32 (K x N) -> bf16 pairs in exact B-fragment order --------
// Sw[(((e*NT + nt)*NK + kb)*32 + lane)*8 + v] = pack(W[k0, n], W[k0+1, n])
//   n = nt*16 + (lane&15), k0 = kb*32 + KB(v, lane>>4)
__global__ void swizzle_w_kernel(const float* __restrict__ W, unsigned int* __restrict__ Sw,
                                 int K, int N) {
  const int gid = blockIdx.x * 256 + threadIdx.x;
  const int v    = gid & 7;
  const int lane = (gid >> 3) & 31;
  const int NK = K / 32;
  const int NT = N / 16;
  const int kb = (gid >> 8) % NK;
  const int nt = ((gid >> 8) / NK) % NT;
  const int e  = (gid >> 8) / (NK * NT);
  const int n  = nt * 16 + (lane & 15);
  const int k0 = kb * 32 + KB(v, lane >> 4);
  const float* We = W + (size_t)e * K * N;
  Sw[gid] = pack_bf16x2(We[(size_t)k0 * N + n], We[(size_t)(k0 + 1) * N + n]);
}

// -------- kernel 4 (fast path): gathered expert MLP, swizzled bf16 weights --------
__global__ void __launch_bounds__(256)
moe_gemm_swz_kernel(const float* __restrict__ x,
                    const unsigned int* __restrict__ Sw1,   // [E][F/16][D/32][32][8]
                    const float* __restrict__ b1,
                    const unsigned int* __restrict__ Sw2,   // [E][D/16][F/32][32][8]
                    const float* __restrict__ b2,
                    const int* __restrict__ list, const int* __restrict__ counts,
                    const float* __restrict__ gval, float* __restrict__ out) {
  __shared__ __align__(16) unsigned int   XL[TILE_M][D / 2];   // x tile, bf16x2 (32 KB)
  __shared__ __align__(16) unsigned short HcL[TILE_M][64];     // gelu(H) chunk, bf16 (4 KB)
  __shared__ int   tokS[TILE_M];
  __shared__ float gtS[TILE_M];

  const int e = blockIdx.y;
  const int mt = blockIdx.x;
  const int cnt = counts[e];
  if (mt * TILE_M >= cnt) return;               // uniform per block

  const int tid = threadIdx.x;
  const int wv = tid >> 5, lane = tid & 31, half = lane >> 4, lrow = lane & 15;

  if (tid < TILE_M) {
    const int i = mt * TILE_M + tid;
    if (i < cnt) { const int s = list[e * T + i]; tokS[tid] = s >> 1; gtS[tid] = gval[s]; }
    else         { tokS[tid] = 0; gtS[tid] = 0.f; }
  }
  __syncthreads();

  { // stage gathered X rows -> bf16 LDS (coalesced float4 loads)
    const int r = tid >> 3, seg = tid & 7;
    const bool valid = (mt * TILE_M + r) < cnt;
    const float4* src = (const float4*)(x + (size_t)tokS[r] * D + seg * 64);
#pragma unroll
    for (int j = 0; j < 16; j++) {
      const float4 v = valid ? src[j] : make_float4(0.f, 0.f, 0.f, 0.f);
      XL[r][seg * 32 + 2 * j]     = pack_bf16x2(v.x, v.y);
      XL[r][seg * 32 + 2 * j + 1] = pack_bf16x2(v.z, v.w);
    }
  }
  __syncthreads();

  const unsigned int* Sw1e = Sw1 + (size_t)e * (F / 16) * (D / 32) * 256;
  const unsigned int* Sw2e = Sw2 + (size_t)e * (D / 16) * (F / 32) * 256;
  const float* b1e = b1 + (size_t)e * F;
  const float* b2e = b2 + (size_t)e * D;

  v8f accD[2][4];
#pragma unroll
  for (int a = 0; a < 2; a++)
#pragma unroll
    for (int b = 0; b < 4; b++) accD[a][b] = {};

  const int rt = wv >> 2, ct = wv & 3;          // this wave's Hc tile in the 32x64 chunk

  for (int fc = 0; fc < F; fc += 64) {
    // ---- stage 1: Hc(16x16) = X(16x512) @ W1(512x16): 16 WMMAs over K=D
    v8f acc1 = {};
    const int ft = (fc >> 4) + ct;              // F column-tile index
    const unsigned int* w1p = Sw1e + ((size_t)(ft * (D / 32)) * 32 + lane) * 8;
#pragma unroll 4
    for (int kbi = 0; kbi < D / 32; kbi++) {
      Frag a, b;
      const uint4* ap = (const uint4*)&XL[rt * 16 + lrow][kbi * 16 + 4 * half];
      a.q[0] = ap[0];
      a.q[1] = ap[2];
      const uint4* bp = (const uint4*)(w1p + (size_t)kbi * 256);
      b.q[0] = bp[0];
      b.q[1] = bp[1];
      acc1 = __builtin_amdgcn_wmma_f32_16x16x32_bf16(false, a.v, false, b.v,
                                                     (short)0, acc1, false, false);
    }
    const int fcol = fc + ct * 16 + lrow;
    const float b1v = b1e[fcol];
#pragma unroll
    for (int r = 0; r < 8; r++) {               // C/D layout: row = r + 8*half, col = lrow
      const float h = acc1[r] + b1v;
      const float g = 0.5f * h * (1.f + erff(h * 0.70710678118f));   // exact GELU
      HcL[rt * 16 + r + 8 * half][ct * 16 + lrow] = bf16_of(g);
    }
    __syncthreads();

    // ---- stage 2: accD(32x512) += gelu(H)(32x64) @ W2[fc:fc+64](64x512)
#pragma unroll
    for (int kc = 0; kc < 64; kc += 32) {
      Frag a0, a1;
      {
        const uint4* p0 = (const uint4*)&HcL[lrow][kc + 8 * half];
        a0.q[0] = p0[0];  a0.q[1] = p0[2];
        const uint4* p1 = (const uint4*)&HcL[16 + lrow][kc + 8 * half];
        a1.q[0] = p1[0];  a1.q[1] = p1[2];
      }
      const int kb2 = (fc + kc) >> 5;           // K-tile index into Sw2
#pragma unroll
      for (int c2 = 0; c2 < 4; c2++) {
        const int dt = wv * 4 + c2;             // D column-tile index
        Frag b;
        const uint4* bp = (const uint4*)(Sw2e + ((size_t)(dt * (F / 32) + kb2) * 32 + lane) * 8);
        b.q[0] = bp[0];
        b.q[1] = bp[1];
        accD[0][c2] = __builtin_amdgcn_wmma_f32_16x16x32_bf16(false, a0.v, false, b.v,
                                                              (short)0, accD[0][c2], false, false);
        accD[1][c2] = __builtin_amdgcn_wmma_f32_16x16x32_bf16(false, a1.v, false, b.v,
                                                              (short)0, accD[1][c2], false, false);
      }
    }
    __syncthreads();
  }

  // ---- gated scatter: exactly 2 contributions per out element -> deterministic
  const int rows = min(TILE_M, cnt - mt * TILE_M);
#pragma unroll
  for (int r2 = 0; r2 < 2; r2++)
#pragma unroll
    for (int c2 = 0; c2 < 4; c2++) {
      const int dcol = wv * 64 + c2 * 16 + lrow;
      const float b2v = b2e[dcol];
#pragma unroll
      for (int r = 0; r < 8; r++) {
        const int m = r2 * 16 + r + 8 * half;
        if (m < rows) {
          const float yv = accD[r2][c2][r] + b2v;
          atomicAdd(&out[(size_t)tokS[m] * D + dcol], gtS[m] * yv);
        }
      }
    }
}

// -------- kernel 4 (fallback): direct f32->bf16 conversion in the inner loop --------
__global__ void __launch_bounds__(256)
moe_gemm_direct_kernel(const float* __restrict__ x,
                       const float* __restrict__ W1, const float* __restrict__ b1,
                       const float* __restrict__ W2, const float* __restrict__ b2,
                       const int* __restrict__ list, const int* __restrict__ counts,
                       const float* __restrict__ gval, float* __restrict__ out) {
  __shared__ __align__(16) unsigned int   XL[TILE_M][D / 2];
  __shared__ __align__(16) unsigned short HcL[TILE_M][64];
  __shared__ int   tokS[TILE_M];
  __shared__ float gtS[TILE_M];

  const int e = blockIdx.y;
  const int mt = blockIdx.x;
  const int cnt = counts[e];
  if (mt * TILE_M >= cnt) return;

  const int tid = threadIdx.x;
  const int wv = tid >> 5, lane = tid & 31, half = lane >> 4, lrow = lane & 15;

  if (tid < TILE_M) {
    const int i = mt * TILE_M + tid;
    if (i < cnt) { const int s = list[e * T + i]; tokS[tid] = s >> 1; gtS[tid] = gval[s]; }
    else         { tokS[tid] = 0; gtS[tid] = 0.f; }
  }
  __syncthreads();

  {
    const int r = tid >> 3, seg = tid & 7;
    const bool valid = (mt * TILE_M + r) < cnt;
    const float4* src = (const float4*)(x + (size_t)tokS[r] * D + seg * 64);
#pragma unroll
    for (int j = 0; j < 16; j++) {
      const float4 v = valid ? src[j] : make_float4(0.f, 0.f, 0.f, 0.f);
      XL[r][seg * 32 + 2 * j]     = pack_bf16x2(v.x, v.y);
      XL[r][seg * 32 + 2 * j + 1] = pack_bf16x2(v.z, v.w);
    }
  }
  __syncthreads();

  const float* W1e = W1 + (size_t)e * D * F;
  const float* W2e = W2 + (size_t)e * F * D;
  const float* b1e = b1 + (size_t)e * F;
  const float* b2e = b2 + (size_t)e * D;

  v8f accD[2][4];
#pragma unroll
  for (int a = 0; a < 2; a++)
#pragma unroll
    for (int b = 0; b < 4; b++) accD[a][b] = {};

  const int rt = wv >> 2, ct = wv & 3;

  for (int fc = 0; fc < F; fc += 64) {
    v8f acc1 = {};
    const int fcol = fc + ct * 16 + lrow;
    for (int kb = 0; kb < D; kb += 32) {
      Frag a, b;
      const uint4* ap = (const uint4*)&XL[rt * 16 + lrow][(kb >> 1) + 4 * half];
      a.q[0] = ap[0];
      a.q[1] = ap[2];
#pragma unroll
      for (int v = 0; v < 8; v++) {
        const int k0 = kb + KB(v, half);
        b.u[v] = pack_bf16x2(W1e[(size_t)k0 * F + fcol], W1e[(size_t)(k0 + 1) * F + fcol]);
      }
      acc1 = __builtin_amdgcn_wmma_f32_16x16x32_bf16(false, a.v, false, b.v,
                                                     (short)0, acc1, false, false);
    }
    const float b1v = b1e[fcol];
#pragma unroll
    for (int r = 0; r < 8; r++) {
      const float h = acc1[r] + b1v;
      const float g = 0.5f * h * (1.f + erff(h * 0.70710678118f));
      HcL[rt * 16 + r + 8 * half][ct * 16 + lrow] = bf16_of(g);
    }
    __syncthreads();

#pragma unroll
    for (int kc = 0; kc < 64; kc += 32) {
      Frag a0, a1;
      const uint4* p0 = (const uint4*)&HcL[lrow][kc + 8 * half];
      a0.q[0] = p0[0];  a0.q[1] = p0[2];
      const uint4* p1 = (const uint4*)&HcL[16 + lrow][kc + 8 * half];
      a1.q[0] = p1[0];  a1.q[1] = p1[2];
#pragma unroll
      for (int c2 = 0; c2 < 4; c2++) {
        const int dcol = wv * 64 + c2 * 16 + lrow;
        Frag b;
#pragma unroll
        for (int v = 0; v < 8; v++) {
          const int fg = fc + kc + KB(v, half);
          b.u[v] = pack_bf16x2(W2e[(size_t)fg * D + dcol], W2e[(size_t)(fg + 1) * D + dcol]);
        }
        accD[0][c2] = __builtin_amdgcn_wmma_f32_16x16x32_bf16(false, a0.v, false, b.v,
                                                              (short)0, accD[0][c2], false, false);
        accD[1][c2] = __builtin_amdgcn_wmma_f32_16x16x32_bf16(false, a1.v, false, b.v,
                                                              (short)0, accD[1][c2], false, false);
      }
    }
    __syncthreads();
  }

  const int rows = min(TILE_M, cnt - mt * TILE_M);
#pragma unroll
  for (int r2 = 0; r2 < 2; r2++)
#pragma unroll
    for (int c2 = 0; c2 < 4; c2++) {
      const int dcol = wv * 64 + c2 * 16 + lrow;
      const float b2v = b2e[dcol];
#pragma unroll
      for (int r = 0; r < 8; r++) {
        const int m = r2 * 16 + r + 8 * half;
        if (m < rows) {
          const float yv = accD[r2][c2][r] + b2v;
          atomicAdd(&out[(size_t)tokS[m] * D + dcol], gtS[m] * yv);
        }
      }
    }
}

extern "C" void kernel_launch(void* const* d_in, const int* in_sizes, int n_in,
                              void* d_out, int out_size, void* d_ws, size_t ws_size,
                              hipStream_t stream) {
  const float* x  = (const float*)d_in[0];
  const float* A  = (const float*)d_in[1];
  const float* W1 = (const float*)d_in[2];
  const float* b1 = (const float*)d_in[3];
  const float* W2 = (const float*)d_in[4];
  const float* b2 = (const float*)d_in[5];

  float* out    = (float*)d_out;                 // (T, D)
  float* an     = out + (size_t)T * D;           // (E, D)
  float* scores = an + (size_t)E * D;            // (T, E)
  float* topkf  = scores + (size_t)T * E;        // (T, 2) indices as float

  char*  ws     = (char*)d_ws;
  int*   eid    = (int*)ws;                                            // 2T ints
  float* gval   = (float*)(ws + (size_t)2 * T * sizeof(int));          // 2T floats
  int*   list   = (int*)(ws + (size_t)2 * T * (sizeof(int) + sizeof(float)));   // E*T ints
  int*   counts = (int*)(ws + (size_t)2 * T * (sizeof(int) + sizeof(float))
                            + (size_t)E * T * sizeof(int));            // E ints

  const size_t BUCKET_BYTES = (size_t)2 * T * 8 + (size_t)E * T * 4 + E * 4;   // 196,640
  const size_t SW_OFF  = (BUCKET_BYTES + 1023) & ~(size_t)1023;
  const size_t SW1_DW  = (size_t)E * (F / 16) * (D / 32) * 256;        // 4,194,304 dwords
  const size_t SW2_DW  = (size_t)E * (D / 16) * (F / 32) * 256;        // 4,194,304 dwords
  const size_t NEED    = SW_OFF + (SW1_DW + SW2_DW) * 4;               // ~33.8 MB

  hipMemsetAsync(out, 0, (size_t)T * D * sizeof(float), stream);
  hipLaunchKernelGGL(anchors_norm_kernel, dim3(1), dim3(256), 0, stream, A, an);
  hipLaunchKernelGGL(route_kernel, dim3(T / 8), dim3(256), 0, stream,
                     x, an, scores, topkf, eid, gval);
  hipLaunchKernelGGL(bucket_kernel, dim3(E), dim3(32), 0, stream, eid, list, counts);

  if (ws_size >= NEED) {
    unsigned int* Sw1 = (unsigned int*)(ws + SW_OFF);
    unsigned int* Sw2 = Sw1 + SW1_DW;
    hipLaunchKernelGGL(swizzle_w_kernel, dim3((unsigned)(SW1_DW / 256)), dim3(256), 0, stream,
                       W1, Sw1, D, F);
    hipLaunchKernelGGL(swizzle_w_kernel, dim3((unsigned)(SW2_DW / 256)), dim3(256), 0, stream,
                       W2, Sw2, F, D);
    hipLaunchKernelGGL(moe_gemm_swz_kernel, dim3(T / TILE_M, E), dim3(256), 0, stream,
                       x, Sw1, b1, Sw2, b2, list, counts, gval, out);
  } else {
    hipLaunchKernelGGL(moe_gemm_direct_kernel, dim3(T / TILE_M, E), dim3(256), 0, stream,
                       x, W1, b1, W2, b2, list, counts, gval, out);
  }
}
